// Block_54400055771799
// MI455X (gfx1250) — matrix operand order repaired and checked by reading, compiled, not run on hardware
//
#include <hip/hip_runtime.h>
#include <hip/hip_bf16.h>
#include <math.h>

typedef _Float16 h4  __attribute__((ext_vector_type(4)));
typedef _Float16 h8  __attribute__((ext_vector_type(8)));
typedef _Float16 h16 __attribute__((ext_vector_type(16)));
typedef float    f8  __attribute__((ext_vector_type(8)));
typedef unsigned int u32x4 __attribute__((ext_vector_type(4)));
typedef int i32x4 __attribute__((ext_vector_type(4)));
typedef int i32x8 __attribute__((ext_vector_type(8)));

#if __has_builtin(__builtin_amdgcn_tensor_load_to_lds)
#define HAVE_TDM 1
#else
#define HAVE_TDM 0
#endif

static __device__ __forceinline__ h16 cat8(h8 lo, h8 hi) {
  h16 r;
#pragma unroll
  for (int i = 0; i < 8; ++i) { r[i] = lo[i]; r[i + 8] = hi[i]; }
  return r;
}

static __device__ __forceinline__ f8 wmma_f16(h16 a, h16 b, f8 c) {
  // v_wmma_f32_16x16x32_f16: (neg_a, A, neg_b, B, c_mod, C, reuse_a, reuse_b)
  return __builtin_amdgcn_wmma_f32_16x16x32_f16(false, a, false, b, (short)0, c,
                                                false, false);
}

// LDS byte address (true AS(3) offset) from a generic pointer into __shared__.
static __device__ __forceinline__ unsigned lds_addr(const void* p) {
  return (unsigned)(unsigned long long)(const __attribute__((address_space(3))) void*)p;
}

// Async DMA: 16 bytes per lane, global -> LDS, tracked by ASYNCcnt.
// GVS addressing: SGPR64 base + VGPR32 byte offset.
static __device__ __forceinline__ void async_b128(unsigned lds, unsigned voff,
                                                  const _Float16* sbase) {
  asm volatile("global_load_async_to_lds_b128 %0, %1, %2"
               :: "v"(lds), "v"(voff), "s"(sbase)
               : "memory");
}

#if HAVE_TDM
// Tensor Data Mover: one instruction DMAs a 2D tile [tile_rows][tile_k] of
// 2-byte elements (row stride = stride_elems) from global into contiguous LDS.
// Descriptor packing per CDNA5 ISA 8.3/8.4 (D# group0 + group1; groups 2/3 zero).
static __device__ __forceinline__ void tdm_load_2d(unsigned lds, const void* gp,
                                                   unsigned tdim0, unsigned tdim1,
                                                   unsigned tile_k, unsigned tile_rows,
                                                   unsigned stride_elems) {
  const unsigned long long ga = (unsigned long long)gp;
  u32x4 g0;
  g0[0] = 1u;                                   // count=1, user mode, no gather
  g0[1] = lds;                                  // lds_addr (bytes)
  g0[2] = (unsigned)ga;                         // global_addr[31:0]
  g0[3] = (unsigned)((ga >> 32) & 0x1FFFFFFu)   // global_addr[56:32]
          | (2u << 30);                         // type = 2 ("image")
  i32x8 g1;
  g1[0] = 0x00010000;                           // wg_mask=0, data_size=1 (2B)
  g1[1] = (int)((tdim0 & 0xFFFFu) << 16);       // tensor_dim0[15:0]
  g1[2] = (int)(((tdim0 >> 16) & 0xFFFFu) | ((tdim1 & 0xFFFFu) << 16));
  g1[3] = (int)(((tdim1 >> 16) & 0xFFFFu) | (tile_k << 16));    // tile_dim0
  g1[4] = (int)(tile_rows & 0xFFFFu);           // tile_dim1 (tile_dim2=0)
  g1[5] = (int)stride_elems;                    // tensor_dim0_stride[31:0]
  g1[6] = 0;                                    // stride[47:32], dim1_stride lo
  g1[7] = 0;
  i32x4 z4 = {0, 0, 0, 0};
#if __clang_major__ >= 23
  i32x8 z8 = {0, 0, 0, 0, 0, 0, 0, 0};
  __builtin_amdgcn_tensor_load_to_lds(g0, g1, z4, z4, z8, 0);
#else
  __builtin_amdgcn_tensor_load_to_lds(g0, g1, z4, z4, 0);
#endif
}
#endif

// ---------------------------------------------------------------------------
// fp32 -> fp16 elementwise convert (x). n must be a multiple of 4.
// ---------------------------------------------------------------------------
__global__ __launch_bounds__(256) void cvt_f16(const float* __restrict__ s,
                                               _Float16* __restrict__ d, int n) {
  int i = (blockIdx.x * 256 + threadIdx.x) * 4;
  if (i >= n) return;
  float4 v = *(const float4*)(s + i);
  h4 o;
  o[0] = (_Float16)v.x; o[1] = (_Float16)v.y;
  o[2] = (_Float16)v.z; o[3] = (_Float16)v.w;
  *(h4*)(d + i) = o;
}

// ---------------------------------------------------------------------------
// fp32 [R][Ncol] -> fp16 transposed [Ncol][R]  (weights: make K contiguous)
// grid = (Ncol/32, R/32), 256 threads.
// ---------------------------------------------------------------------------
__global__ __launch_bounds__(256) void cvt_tr(const float* __restrict__ s,
                                              _Float16* __restrict__ d,
                                              int R, int Ncol) {
  __shared__ float t[32][33];
  const int tx = threadIdx.x & 31, ty = threadIdx.x >> 5;
  const int c = blockIdx.x * 32 + tx;
  const int r0 = blockIdx.y * 32 + ty;
#pragma unroll
  for (int j = 0; j < 32; j += 8) t[ty + j][tx] = s[(size_t)(r0 + j) * Ncol + c];
  __syncthreads();
  const int c2 = blockIdx.y * 32 + tx;  // row index of src
  const int r2 = blockIdx.x * 32 + ty;  // col index of src
#pragma unroll
  for (int j = 0; j < 32; j += 8)
    d[(size_t)(r2 + j) * R + c2] = (_Float16)t[tx][ty + j];
}

// ---------------------------------------------------------------------------
// WMMA GEMM: out = A[M][K](f16) * Bt[N][K](f16)^T  (+bias +resid, relu)
// 128x128 block tile, 256 threads = 8 waves (4x2), K-step 32.
// Double-buffered LDS staging via the Tensor Data Mover (TENSORcnt); async
// b128 fallback if the TDM builtin is unavailable.
// ---------------------------------------------------------------------------
__global__ __launch_bounds__(256) void gemm_wmma(
    const _Float16* __restrict__ A, const _Float16* __restrict__ Bt,
    const float* __restrict__ bias, const float* __restrict__ resid,
    float* __restrict__ outF, _Float16* __restrict__ outH,
    int M, int N, int K, int relu) {
  __shared__ _Float16 As[2][128 * 32];
  __shared__ _Float16 Bs[2][128 * 32];
  const int tid = threadIdx.x;
  const int lh = (tid >> 4) & 1, ln = tid & 15;
  const int wave = tid >> 5;
  const int wm = wave >> 1, wn = wave & 1;  // 4x2 wave grid
  const int m0 = blockIdx.x * 128, n0 = blockIdx.y * 128;

#if HAVE_TDM
  const bool issuer = (tid < 32);  // wave 0 drives the TDM
  const unsigned ldsA[2] = {lds_addr(&As[0][0]), lds_addr(&As[1][0])};
  const unsigned ldsB[2] = {lds_addr(&Bs[0][0]), lds_addr(&Bs[1][0])};
  auto issue_tile = [&](int buf, int k0) {
    if (issuer) {
      tdm_load_2d(ldsA[buf], A + (size_t)m0 * K + k0, (unsigned)K, 128u, 32u,
                  128u, (unsigned)K);
      tdm_load_2d(ldsB[buf], Bt + (size_t)n0 * K + k0, (unsigned)K, 128u, 32u,
                  128u, (unsigned)K);
    }
  };
#else
  // staging: thread owns 32B of A and 32B of B per K-step (2 async b128 each)
  const int sRow = tid >> 1;
  const int sCol = (tid & 1) << 4;  // halves
  const unsigned gA0 = (unsigned)(((size_t)(m0 + sRow) * K + sCol) * 2);
  const unsigned gB0 = (unsigned)(((size_t)(n0 + sRow) * K + sCol) * 2);
  const unsigned la[2] = {lds_addr(&As[0][sRow * 32 + sCol]),
                          lds_addr(&As[1][sRow * 32 + sCol])};
  const unsigned lb[2] = {lds_addr(&Bs[0][sRow * 32 + sCol]),
                          lds_addr(&Bs[1][sRow * 32 + sCol])};
  auto issue_tile = [&](int buf, int k0) {
    const unsigned kb = (unsigned)(k0 * 2);  // byte offset along K
    async_b128(la[buf],      gA0 + kb,      A);
    async_b128(la[buf] + 16, gA0 + kb + 16, A);
    async_b128(lb[buf],      gB0 + kb,      Bt);
    async_b128(lb[buf] + 16, gB0 + kb + 16, Bt);
  };
#endif

  f8 zero = {};
  f8 acc[2][4];
#pragma unroll
  for (int i = 0; i < 2; ++i)
#pragma unroll
    for (int j = 0; j < 4; ++j) acc[i][j] = zero;

  const int nk = K >> 5;
  issue_tile(0, 0);
  for (int it = 0; it < nk; ++it) {
    const int cur = it & 1;
    if (it + 1 < nk) {
      issue_tile(cur ^ 1, (it + 1) << 5);  // overlap next slab with compute
#if HAVE_TDM
      __builtin_amdgcn_s_wait_tensorcnt(2);  // current buffer's 2 DMAs landed
#else
      asm volatile("s_wait_asynccnt 0x4" ::: "memory");
#endif
    } else {
#if HAVE_TDM
      __builtin_amdgcn_s_wait_tensorcnt(0);
#else
      asm volatile("s_wait_asynccnt 0x0" ::: "memory");
#endif
    }
    __syncthreads();
    h16 af[2], bf[4];
#pragma unroll
    for (int tm = 0; tm < 2; ++tm) {
      const _Float16* ap = &As[cur][(wm * 32 + tm * 16 + ln) * 32 + lh * 8];
      af[tm] = cat8(*(const h8*)ap, *(const h8*)(ap + 16));
    }
#pragma unroll
    for (int tn = 0; tn < 4; ++tn)
      bf[tn] = cat8(*(const h8*)&Bs[cur][(wn * 64 + tn * 16 + ln) * 32 + lh * 8],
                    *(const h8*)&Bs[cur][(wn * 64 + tn * 16 + ln) * 32 + lh * 8 + 16]);
#pragma unroll
    for (int tm = 0; tm < 2; ++tm)
#pragma unroll
      for (int tn = 0; tn < 4; ++tn)
        acc[tm][tn] = wmma_f16(af[tm], bf[tn], acc[tm][tn]);
    __syncthreads();
  }

#pragma unroll
  for (int tm = 0; tm < 2; ++tm)
#pragma unroll
    for (int tn = 0; tn < 4; ++tn) {
      const int col = n0 + wn * 64 + tn * 16 + ln;
      const float bv = bias ? bias[col] : 0.0f;
#pragma unroll
      for (int r = 0; r < 8; ++r) {
        const int row = m0 + wm * 32 + tm * 16 + r + 8 * lh;
        float v = acc[tm][tn][r] + bv;
        if (resid) v += resid[(size_t)row * N + col];
        if (relu) v = fmaxf(v, 0.0f);
        if (outF) outF[(size_t)row * N + col] = v;
        if (outH) outH[(size_t)row * N + col] = (_Float16)v;
      }
    }
}

// ---------------------------------------------------------------------------
// Flash attention (causal). q,k,v,o: f16 [B*T][C], head h at cols h*64..h*64+63.
// grid = (T/64, B*H), 128 threads = 4 waves; wave owns 16 q-rows.
// K-tile staged via async-to-LDS; V staged transposed manually.
// ---------------------------------------------------------------------------
__global__ __launch_bounds__(128) void attn_wmma(
    const _Float16* __restrict__ q, const _Float16* __restrict__ k,
    const _Float16* __restrict__ v, _Float16* __restrict__ o,
    int T, int C, int H) {
  __shared__ _Float16 Ks[32 * 64];      // [key][d]
  __shared__ _Float16 Vt[64 * 32];      // [d][key]
  __shared__ _Float16 Pb[4 * 16 * 32];  // per-wave P re-layout buffer
  const int tid = threadIdx.x;
  const int wave = tid >> 5;
  const int lh = (tid >> 4) & 1, ln = tid & 15;
  const int qt = blockIdx.x;
  const int bh = blockIdx.y;
  const int b = bh / H, h = bh % H;
  const size_t base = (size_t)b * T * C + (size_t)h * 64;
  const float scale = 1.0f / sqrtf((float)C);

  // q A-fragments (K-dim d = 64 -> two 16x32 fragments), loaded once
  const int qrow = qt * 64 + wave * 16 + ln;  // lane's A-matrix row (M = ln)
  h16 qf[2];
#pragma unroll
  for (int dc = 0; dc < 2; ++dc) {
    const _Float16* qp = q + base + (size_t)qrow * C + dc * 32 + lh * 8;
    qf[dc] = cat8(*(const h8*)qp, *(const h8*)(qp + 16));
  }

  f8 zero = {};
  f8 O[4];
#pragma unroll
  for (int i = 0; i < 4; ++i) O[i] = zero;
  float mrow[8], lrow[8];
#pragma unroll
  for (int r = 0; r < 8; ++r) { mrow[r] = -INFINITY; lrow[r] = 0.0f; }

  const int sr = tid >> 2;          // staging: key row 0..31
  const int sc = (tid & 3) << 4;    // staging: d col 0,16,32,48
  const unsigned ksl = lds_addr(Ks + sr * 64 + sc);
  const int kend = qt * 64 + 64;    // causal bound for this q-tile

  for (int kb = 0; kb < kend; kb += 32) {
    // stage K tile [32][64] via async DMA, V tile transposed [64][32] manually
    const unsigned kvo = (unsigned)(((size_t)(kb + sr) * C + sc) * 2);
    async_b128(ksl,      kvo,      k + base);
    async_b128(ksl + 16, kvo + 16, k + base);
    h16 vv = *(const h16*)(v + base + (size_t)(kb + sr) * C + sc);
#pragma unroll
    for (int i = 0; i < 16; ++i) Vt[(sc + i) * 32 + sr] = vv[i];
    asm volatile("s_wait_asynccnt 0x0" ::: "memory");
    __syncthreads();

    // S = q @ k^T : two 16x16 tiles (kn), K-dim = 64 (two WMMAs each)
    f8 s[2];
#pragma unroll
    for (int kn = 0; kn < 2; ++kn) {
      f8 c = zero;
#pragma unroll
      for (int dc = 0; dc < 2; ++dc) {
        h16 kf = cat8(*(const h8*)(Ks + (kn * 16 + ln) * 64 + dc * 32 + lh * 8),
                      *(const h8*)(Ks + (kn * 16 + ln) * 64 + dc * 32 + lh * 8 + 16));
        c = wmma_f16(qf[dc], kf, c);
      }
      s[kn] = c;
    }
    // scale + causal mask (D layout: lane holds N=ln, rows r+8*lh)
#pragma unroll
    for (int kn = 0; kn < 2; ++kn)
#pragma unroll
      for (int r = 0; r < 8; ++r) {
        const int kg = kb + kn * 16 + ln;
        const int qg = qt * 64 + wave * 16 + r + 8 * lh;
        float val = s[kn][r] * scale;
        s[kn][r] = (kg <= qg) ? val : -INFINITY;
      }
    // online softmax: row max / correction / exp / row sum
    float corr[8];
#pragma unroll
    for (int r = 0; r < 8; ++r) {
      float rm = fmaxf(s[0][r], s[1][r]);
#pragma unroll
      for (int msk = 1; msk <= 8; msk <<= 1) rm = fmaxf(rm, __shfl_xor(rm, msk, 32));
      const float mn = fmaxf(mrow[r], rm);
      corr[r] = __expf(mrow[r] - mn);
      mrow[r] = mn;
    }
#pragma unroll
    for (int r = 0; r < 8; ++r) {
      float rs = 0.0f;
#pragma unroll
      for (int kn = 0; kn < 2; ++kn) {
        const float p = __expf(s[kn][r] - mrow[r]);
        rs += p;
        Pb[wave * 512 + (r + 8 * lh) * 32 + kn * 16 + ln] = (_Float16)p;
      }
#pragma unroll
      for (int msk = 1; msk <= 8; msk <<= 1) rs += __shfl_xor(rs, msk, 32);
      lrow[r] = lrow[r] * corr[r] + rs;
#pragma unroll
      for (int nd = 0; nd < 4; ++nd) O[nd][r] *= corr[r];
    }
    // O += P @ V  (P re-layout D->A through LDS; same-wave LDS is in-order)
    const _Float16* pp = Pb + wave * 512 + ln * 32 + lh * 8;
    h16 pf = cat8(*(const h8*)pp, *(const h8*)(pp + 16));
#pragma unroll
    for (int nd = 0; nd < 4; ++nd) {
      h16 vf = cat8(*(const h8*)(Vt + (nd * 16 + ln) * 32 + lh * 8),
                    *(const h8*)(Vt + (nd * 16 + ln) * 32 + lh * 8 + 16));
      O[nd] = wmma_f16(pf, vf, O[nd]);
    }
    __syncthreads();
  }

  // normalize and write attention output (f16)
#pragma unroll
  for (int nd = 0; nd < 4; ++nd)
#pragma unroll
    for (int r = 0; r < 8; ++r) {
      const int row = qt * 64 + wave * 16 + r + 8 * lh;
      const int col = nd * 16 + ln;
      o[base + (size_t)row * C + col] = (_Float16)(O[nd][r] / lrow[r]);
    }
}

// ---------------------------------------------------------------------------
// Launch: full transformer block
// ---------------------------------------------------------------------------
extern "C" void kernel_launch(void* const* d_in, const int* in_sizes, int n_in,
                              void* d_out, int out_size, void* d_ws, size_t ws_size,
                              hipStream_t stream) {
  (void)in_sizes; (void)n_in; (void)out_size; (void)ws_size;
  constexpr int Bc = 2, T = 2048, C = 1024, H = 16;
  constexpr int M = Bc * T;      // 4096
  constexpr int F = 4 * C;       // 4096

  const float* x     = (const float*)d_in[0];
  const float* Wq    = (const float*)d_in[1];
  const float* Wk    = (const float*)d_in[2];
  const float* Wv    = (const float*)d_in[3];
  const float* Wp    = (const float*)d_in[4];
  const float* bproj = (const float*)d_in[5];
  const float* W1    = (const float*)d_in[6];
  const float* b1    = (const float*)d_in[7];
  const float* W2    = (const float*)d_in[8];
  const float* b2    = (const float*)d_in[9];

  char* p = (char*)d_ws;
  auto take = [&](size_t bytes) {
    char* r = p;
    p += (bytes + 255) & ~(size_t)255;
    return (void*)r;
  };
  _Float16* xh  = (_Float16*)take((size_t)M * C * 2);
  _Float16* wqt = (_Float16*)take((size_t)C * C * 2);
  _Float16* wkt = (_Float16*)take((size_t)C * C * 2);
  _Float16* wvt = (_Float16*)take((size_t)C * C * 2);
  _Float16* wpt = (_Float16*)take((size_t)C * C * 2);
  _Float16* w1t = (_Float16*)take((size_t)C * F * 2);   // [F][C]
  _Float16* w2t = (_Float16*)take((size_t)F * C * 2);   // [C][F]
  _Float16* qh  = (_Float16*)take((size_t)M * C * 2);
  _Float16* kh  = (_Float16*)take((size_t)M * C * 2);
  _Float16* vh  = (_Float16*)take((size_t)M * C * 2);
  _Float16* ah  = (_Float16*)take((size_t)M * C * 2);
  float*    y1  = (float*)take((size_t)M * C * 4);
  _Float16* y1h = (_Float16*)take((size_t)M * C * 2);
  _Float16* h1h = (_Float16*)take((size_t)M * F * 2);

  // conversions
  cvt_f16<<<(M * C) / 1024, 256, 0, stream>>>(x, xh, M * C);
  cvt_tr<<<dim3(C / 32, C / 32), 256, 0, stream>>>(Wq, wqt, C, C);
  cvt_tr<<<dim3(C / 32, C / 32), 256, 0, stream>>>(Wk, wkt, C, C);
  cvt_tr<<<dim3(C / 32, C / 32), 256, 0, stream>>>(Wv, wvt, C, C);
  cvt_tr<<<dim3(C / 32, C / 32), 256, 0, stream>>>(Wp, wpt, C, C);
  cvt_tr<<<dim3(F / 32, C / 32), 256, 0, stream>>>(W1, w1t, C, F);
  cvt_tr<<<dim3(C / 32, F / 32), 256, 0, stream>>>(W2, w2t, F, C);

  const dim3 gCC(M / 128, C / 128);
  // QKV projections
  gemm_wmma<<<gCC, 256, 0, stream>>>(xh, wqt, nullptr, nullptr, nullptr, qh, M, C, C, 0);
  gemm_wmma<<<gCC, 256, 0, stream>>>(xh, wkt, nullptr, nullptr, nullptr, kh, M, C, C, 0);
  gemm_wmma<<<gCC, 256, 0, stream>>>(xh, wvt, nullptr, nullptr, nullptr, vh, M, C, C, 0);

  // causal flash attention
  attn_wmma<<<dim3(T / 64, Bc * H), 128, 0, stream>>>(qh, kh, vh, ah, T, C, H);

  // y1 = x + attn @ Wproj + bproj  (f32 + f16 copies)
  gemm_wmma<<<gCC, 256, 0, stream>>>(ah, wpt, bproj, x, y1, y1h, M, C, C, 0);
  // h1 = relu(y1 @ W1 + b1)
  gemm_wmma<<<dim3(M / 128, F / 128), 256, 0, stream>>>(y1h, w1t, b1, nullptr,
                                                        nullptr, h1h, M, F, C, 1);
  // out = y1 + h1 @ W2 + b2
  gemm_wmma<<<gCC, 256, 0, stream>>>(h1h, w2t, b2, y1, (float*)d_out, nullptr,
                                     M, C, F, 0);
}
